// MultiHeadAttention_20847771255019
// MI455X (gfx1250) — matrix operand order repaired and checked by reading, not compiled, not run on hardware
//
#include <hip/hip_runtime.h>
#include <hip/hip_bf16.h>
#include <stdint.h>

// Problem dims (compile-time, from reference)
#define EMBED 1024
#define HEADS 16
#define HDIM  64
#define SEQ   2048
#define BATCH 2
#define E3    (3 * EMBED)
#define MROWS (BATCH * SEQ)   // 4096

#define LDA 40   // LDS row stride in halves: 80B rows -> 16B-aligned, 20-dword
                 // bank stride -> 16 distinct banks over a fragment's 16 lanes

typedef _Float16 v16h __attribute__((ext_vector_type(16)));
typedef _Float16 v8h  __attribute__((ext_vector_type(8)));
typedef _Float16 v4h  __attribute__((ext_vector_type(4)));
typedef float    v8f  __attribute__((ext_vector_type(8)));
typedef float    v4f  __attribute__((ext_vector_type(4)));

union AF {
    v16h     v;
    v8h      h8[2];
    uint32_t u[8];
};

// ISA 7.12.2 fragment layouts, re-expressed as contiguous 16B spans per lane:
//  A 16x32 (lane m = ln, half hi): halves [8*hi .. 8*hi+7] and [16+8*hi .. 23+8*hi]
//  B 32x16 (lane n = ln, half hi): halves [8*hi .. 8*hi+7] and [16+8*hi .. 23+8*hi]
__device__ __forceinline__ AF load_a_frag(const _Float16* row, int hi) {
    AF f;
    f.h8[0] = *(const v8h*)(row + 8 * hi);
    f.h8[1] = *(const v8h*)(row + 16 + 8 * hi);
    return f;
}
__device__ __forceinline__ AF load_b_frag(const _Float16* row, int hi) {
    AF f;
    f.h8[0] = *(const v8h*)(row + 8 * hi);
    f.h8[1] = *(const v8h*)(row + 16 + 8 * hi);
    return f;
}

__device__ __forceinline__ v8f wmma_f16(const AF& a, const AF& b, v8f c) {
    return __builtin_amdgcn_wmma_f32_16x16x32_f16(
        /*neg_a=*/false, a.v, /*neg_b=*/false, b.v,
        /*c_mod=*/(short)0, c, /*reuse_a=*/false, /*reuse_b=*/false);
}

// ---------------------------------------------------------------------------
// GEMM: C[M,N] = A[M,K] * B[K,N] + bias[N]
// f32 global inputs, f16 WMMA with f32 accumulate.
// Block: 128 threads (4 waves), 64(M) x 128(N) tile; wave tile 32x64
// (2 A-frags x 4 B-frags -> 8 wmma per K-step of 32).
// Requires M%64==0, N%128==0, K%32==0 (4096 x {3072,1024} x 1024: OK).
// ---------------------------------------------------------------------------
template <bool F16OUT>
__global__ __launch_bounds__(128)
void gemm_wmma_kernel(const float* __restrict__ A, const float* __restrict__ B,
                      const float* __restrict__ bias, void* __restrict__ Cout,
                      int M, int N, int K) {
    __shared__ __align__(16) _Float16 sA[64 * LDA];    // [m][k] row-major
    __shared__ __align__(16) _Float16 sB[128 * LDA];   // [n][k] N-major (B^T)

    const int tid  = threadIdx.x;
    const int lane = tid & 31;
    const int wave = tid >> 5;
    const int hi   = lane >> 4;
    const int ln   = lane & 15;
    const int wm   = (wave & 1) * 32;
    const int wn   = (wave >> 1) * 64;
    const int m0   = blockIdx.x * 64;
    const int n0   = blockIdx.y * 128;

    v8f acc[2][4] = {};

    for (int k0 = 0; k0 < K; k0 += 32) {
        __syncthreads();   // protect previous iteration's fragment reads
        // Stage A 64x32: float4 global loads (b128), v4h LDS stores (b64)
        #pragma unroll
        for (int i = tid; i < 64 * 8; i += 128) {
            const int m = i >> 3, kc = (i & 7) * 4;
            const v4f f = *(const v4f*)&A[(size_t)(m0 + m) * K + k0 + kc];
            v4h h; h[0] = (_Float16)f[0]; h[1] = (_Float16)f[1];
            h[2] = (_Float16)f[2]; h[3] = (_Float16)f[3];
            *(v4h*)(sA + m * LDA + kc) = h;
        }
        // Stage B 32x128 transposed to [n][k]: float4 loads, b16 scatter
        #pragma unroll
        for (int i = tid; i < 32 * 32; i += 128) {
            const int kc = i >> 5, n4 = (i & 31) * 4;
            const v4f f = *(const v4f*)&B[(size_t)(k0 + kc) * N + n0 + n4];
            #pragma unroll
            for (int e = 0; e < 4; ++e)
                sB[(n4 + e) * LDA + kc] = (_Float16)f[e];
        }
        // Prefetch next K-tile while this tile computes (global_prefetch_b8)
        if (k0 + 32 < K) {
            __builtin_prefetch(&A[(size_t)(m0 + (tid >> 3)) * K + k0 + 32 + (tid & 7) * 4], 0, 0);
            __builtin_prefetch(&B[(size_t)(k0 + 32 + (tid >> 5)) * N + n0 + (tid & 31) * 4], 0, 0);
        }
        __syncthreads();

        AF a[2], b[4];
        #pragma unroll
        for (int t = 0; t < 2; ++t)
            a[t] = load_a_frag(sA + (wm + t * 16 + ln) * LDA, hi);
        #pragma unroll
        for (int u = 0; u < 4; ++u)
            b[u] = load_b_frag(sB + (wn + u * 16 + ln) * LDA, hi);
        #pragma unroll
        for (int t = 0; t < 2; ++t)
            #pragma unroll
            for (int u = 0; u < 4; ++u)
                acc[t][u] = wmma_f16(a[t], b[u], acc[t][u]);
    }

    // Epilogue: C layout — VGPR r: (m = r + 8*hi, n = ln)
    #pragma unroll
    for (int t = 0; t < 2; ++t) {
        #pragma unroll
        for (int u = 0; u < 4; ++u) {
            const int n  = n0 + wn + u * 16 + ln;
            const float bv = bias[n];
            #pragma unroll
            for (int r = 0; r < 8; ++r) {
                const int m = m0 + wm + t * 16 + r + hi * 8;
                const float val = acc[t][u][r] + bv;
                if (F16OUT)
                    ((_Float16*)Cout)[(size_t)m * N + n] = (_Float16)val;
                else
                    ((float*)Cout)[(size_t)m * N + n] = val;
            }
        }
    }
}

// ---------------------------------------------------------------------------
// Causal flash attention over the f16 qkv workspace [B,S,3E].
// Grid: (S/64, HEADS, BATCH), block 128 (4 waves); wave w owns 16 query rows.
// Online softmax; scores/PV via v_wmma_f32_16x16x32_f16.
// ---------------------------------------------------------------------------
__global__ __launch_bounds__(128)
void flash_attn_kernel(const _Float16* __restrict__ qkv,
                       float* __restrict__ attn) {
    // sK stride 72 halves = 144B rows: 16B-aligned for v8h ops, 36-dword
    // bank stride -> conflict-free B-fragment reads.
    __shared__ __align__(16) _Float16 sK[32 * 72];         // [j][d]
    __shared__ __align__(16) _Float16 sV[64 * LDA];        // [d][j] (V^T)
    __shared__ __align__(16) _Float16 sP[4][16 * LDA];     // per-wave P tile

    const int tid  = threadIdx.x;
    const int lane = tid & 31;
    const int wave = tid >> 5;
    const int hi   = lane >> 4;
    const int ln   = lane & 15;
    const int it   = blockIdx.x;
    const int h    = blockIdx.y;
    const int b    = blockIdx.z;
    const int qi0  = it * 64 + wave * 16;

    const size_t base = (size_t)b * SEQ * E3;
    const int qcol = h * HDIM;
    const int kcol = EMBED + h * HDIM;
    const int vcol = 2 * EMBED + h * HDIM;

    // Q fragments straight from global (row layout == A-frag layout): b128s.
    AF qf[2];
    {
        const _Float16* qrow = qkv + base + (size_t)(qi0 + ln) * E3 + qcol;
        qf[0] = load_a_frag(qrow, hi);
        qf[1] = load_a_frag(qrow + 32, hi);
    }

    v8f acc[4] = {};                    // 16 rows x 64 dims, f32
    float mrow[8], lrow[8];
    #pragma unroll
    for (int r = 0; r < 8; ++r) { mrow[r] = -3.0e38f; lrow[r] = 0.0f; }

    const int jend = it * 64 + 64;      // causal bound for this block
    for (int j0 = 0; j0 < jend; j0 += 32) {
        __syncthreads();
        // Stage K tile [32][64]: 16B global loads + 16B LDS stores
        #pragma unroll
        for (int i = tid; i < 32 * 8; i += 128) {
            const int j = i >> 3, d8 = (i & 7) * 8;
            *(v8h*)(sK + j * 72 + d8) =
                *(const v8h*)(qkv + base + (size_t)(j0 + j) * E3 + kcol + d8);
        }
        // Stage V^T [64][32]: 16B global loads, b16 transpose scatter
        #pragma unroll
        for (int i = tid; i < 32 * 8; i += 128) {
            const int j = i >> 3, d8 = (i & 7) * 8;
            const v8h t =
                *(const v8h*)(qkv + base + (size_t)(j0 + j) * E3 + vcol + d8);
            #pragma unroll
            for (int e = 0; e < 8; ++e)
                sV[(d8 + e) * LDA + j] = t[e];
        }
        if (j0 + 32 < jend) {   // prefetch next K/V tile
            const int j = tid >> 3, d8 = (tid & 7) * 8;
            __builtin_prefetch(qkv + base + (size_t)(j0 + 32 + j) * E3 + kcol + d8, 0, 0);
            __builtin_prefetch(qkv + base + (size_t)(j0 + 32 + j) * E3 + vcol + d8, 0, 0);
        }
        __syncthreads();

        // S = Q K^T: load all four K B-frags first (ILP), then 4 wmma.
        AF kb[2][2];
        #pragma unroll
        for (int js = 0; js < 2; ++js) {
            const _Float16* rowK = sK + (js * 16 + ln) * 72;
            kb[js][0] = load_b_frag(rowK, hi);        // d 0..31
            kb[js][1] = load_b_frag(rowK + 32, hi);   // d 32..63
        }
        v8f s[2];
        #pragma unroll
        for (int js = 0; js < 2; ++js) {
            v8f z = {};
            z = wmma_f16(qf[0], kb[js][0], z);
            z = wmma_f16(qf[1], kb[js][1], z);
            s[js] = z;
        }

        // Online softmax. Row r lives in one 16-lane half: reduce over lanes.
        float alpha[8];
        #pragma unroll
        for (int r = 0; r < 8; ++r) {
            const int iabs = qi0 + r + hi * 8;
            float s0 = s[0][r] * 0.125f;               // 1/sqrt(64)
            float s1 = s[1][r] * 0.125f;
            if (j0 + ln > iabs)      s0 = -3.0e38f;    // causal mask
            if (j0 + 16 + ln > iabs) s1 = -3.0e38f;
            float mc = fmaxf(s0, s1);
            #pragma unroll
            for (int msk = 1; msk < 16; msk <<= 1)
                mc = fmaxf(mc, __shfl_xor(mc, msk, 32));
            const float mn = fmaxf(mrow[r], mc);
            const float al = __expf(mrow[r] - mn);
            const float p0 = __expf(s0 - mn);
            const float p1 = __expf(s1 - mn);
            float rs = p0 + p1;
            #pragma unroll
            for (int msk = 1; msk < 16; msk <<= 1)
                rs += __shfl_xor(rs, msk, 32);
            lrow[r]  = lrow[r] * al + rs;
            mrow[r]  = mn;
            alpha[r] = al;
            // C-layout -> LDS so we can re-read in A-fragment layout
            sP[wave][(r + hi * 8) * LDA + ln]      = (_Float16)p0;
            sP[wave][(r + hi * 8) * LDA + 16 + ln] = (_Float16)p1;
        }
        __syncthreads();   // uniform across waves; publishes sP

        #pragma unroll
        for (int d = 0; d < 4; ++d)
            #pragma unroll
            for (int r = 0; r < 8; ++r)
                acc[d][r] *= alpha[r];

        // P as A-fragment (16x32), V as four B-fragments; acc += P V
        const AF pf = load_a_frag(&sP[wave][ln * LDA], hi);
        AF vb[4];
        #pragma unroll
        for (int d = 0; d < 4; ++d)
            vb[d] = load_b_frag(sV + (d * 16 + ln) * LDA, hi);
        #pragma unroll
        for (int d = 0; d < 4; ++d)
            acc[d] = wmma_f16(pf, vb[d], acc[d]);
    }

    // Normalize and store f32 attention output [B,S,E] (e = h*64 + d)
    #pragma unroll
    for (int r = 0; r < 8; ++r) {
        const float inv = 1.0f / lrow[r];
        const int m = qi0 + r + hi * 8;
        float* orow = attn + ((size_t)b * SEQ + m) * EMBED + h * HDIM;
        #pragma unroll
        for (int d = 0; d < 4; ++d)
            orow[d * 16 + ln] = acc[d][r] * inv;
    }
}

// ---------------------------------------------------------------------------
extern "C" void kernel_launch(void* const* d_in, const int* in_sizes, int n_in,
                              void* d_out, int out_size, void* d_ws, size_t ws_size,
                              hipStream_t stream) {
    (void)in_sizes; (void)n_in; (void)out_size; (void)ws_size;
    const float* x    = (const float*)d_in[0];   // [B,S,E]
    const float* Wqkv = (const float*)d_in[1];   // [E,3E]
    const float* bqkv = (const float*)d_in[2];   // [3E]
    const float* Wout = (const float*)d_in[3];   // [E,E]
    const float* bout = (const float*)d_in[4];   // [E]

    // Workspace: qkv f16 [B,S,3E] (25.2 MB) + attn f32 [B,S,E] (16.8 MB)
    _Float16* qkv = (_Float16*)d_ws;
    float* attn = (float*)((char*)d_ws + (size_t)MROWS * E3 * sizeof(_Float16));

    const dim3 blk(128);

    // 1) qkv = x @ W_qkv + b_qkv  -> f16 workspace
    gemm_wmma_kernel<true><<<dim3(MROWS / 64, E3 / 128), blk, 0, stream>>>(
        x, Wqkv, bqkv, qkv, MROWS, E3, EMBED);

    // 2) causal flash attention -> f32 workspace [B,S,E]
    flash_attn_kernel<<<dim3(SEQ / 64, HEADS, BATCH), blk, 0, stream>>>(qkv, attn);

    // 3) out = attn @ W_out + b_out -> f32 d_out
    gemm_wmma_kernel<false><<<dim3(MROWS / 64, EMBED / 128), blk, 0, stream>>>(
        attn, Wout, bout, d_out, MROWS, EMBED, EMBED);
}
